// DFlashModel_40441412059537
// MI455X (gfx1250) — matrix operand order repaired and checked by reading, not compiled, not run on hardware
//
#include <hip/hip_runtime.h>
#include <hip/hip_bf16.h>

// ---------------------------------------------------------------------------
// DFlash draft-loss pipeline for MI455X (gfx1250, wave32, WMMA).
//
// Roofline: logits GEMM (4096x1024 @ 1024x32000 = 268 GFLOP) dominates ->
// bf16 WMMA with fp32 accumulate. lm_head in bf16 = 65.5MB, fits the 192MB L2,
// so all 256 row-blocks stream it at L2 bandwidth. Logits are never stored:
// online logsumexp + target-logit capture fused into the GEMM epilogue.
//
// Data movement: all GEMM weights are stored pre-transposed (NxK) in bf16 so
// both A and B fragments are contiguous; LDS tiles are double-buffered and
// filled with GLOBAL_LOAD_ASYNC_TO_LDS_B128 (ASYNCcnt), overlapping the next
// tile's fill with the current tile's WMMAs.
// ---------------------------------------------------------------------------

#define B_  2
#define S_  2048
#define D_  1024
#define H_  8
#define V_  32000
#define N_  128
#define BS_ 16
#define HD_ 128
#define NEGF (-1.0e9f)

typedef __bf16 bf16_t;
typedef __attribute__((ext_vector_type(16))) __bf16 v16bf;
typedef __attribute__((ext_vector_type(8)))  __bf16 v8bf;
typedef __attribute__((ext_vector_type(8)))  float  v8f;
typedef int v4i __attribute__((vector_size(16)));

union bfu { v16bf v; v8bf h[2]; };

#define AS1 __attribute__((address_space(1)))
#define AS3 __attribute__((address_space(3)))

// CDNA5 async global->LDS copy (ASYNCcnt) with synchronous fallback.
#if __has_builtin(__builtin_amdgcn_global_load_async_to_lds_b128)
#define HAVE_ASYNC_LDS 1
__device__ __forceinline__ void async_copy_b128(const void* gsrc, void* ldst) {
  __builtin_amdgcn_global_load_async_to_lds_b128((AS1 v4i*)gsrc, (AS3 v4i*)ldst, 0, 0);
}
#if __has_builtin(__builtin_amdgcn_s_wait_asynccnt)
#define WAIT_ASYNC(n) __builtin_amdgcn_s_wait_asynccnt(n)
#else
#define WAIT_ASYNC(n) asm volatile("s_wait_asynccnt %0" ::"i"(n) : "memory")
#endif
#else
#define HAVE_ASYNC_LDS 0
__device__ __forceinline__ void async_copy_b128(const void* gsrc, void* ldst) {
  *(v8bf*)ldst = *(const v8bf*)gsrc;
}
#define WAIT_ASYNC(n) ((void)0)
#endif

__device__ __forceinline__ v8f wmma_bf16(v16bf a, v16bf b, v8f c) {
  // (neg_a, A, neg_b, B, c_mod, C, reuse_a, reuse_b)
  return __builtin_amdgcn_wmma_f32_16x16x32_bf16(false, a, false, b, (short)0, c,
                                                 false, false);
}

// ---------------------------------------------------------------------------
// Elementwise helpers
// ---------------------------------------------------------------------------
__global__ __launch_bounds__(256)
void convert_bf16_kernel(const float* __restrict__ in, bf16_t* __restrict__ out, long n) {
  long i = (long)blockIdx.x * 256 + threadIdx.x;
  if (i < n) out[i] = (bf16_t)in[i];
}

// (K,Nn) fp32 row-major -> (Nn,K) bf16 row-major (weights pre-transposed for GEMM B).
__global__ __launch_bounds__(256)
void convert_T_bf16_kernel(const float* __restrict__ in, bf16_t* __restrict__ out,
                           int K, int Nn) {
  long i = (long)blockIdx.x * 256 + threadIdx.x;
  long total = (long)K * Nn;
  if (i >= total) return;
  int n = (int)(i % Nn);
  int k = (int)(i / Nn);
  out[(size_t)n * K + k] = (bf16_t)in[i];
}

__global__ __launch_bounds__(256)
void concat_hs_kernel(const float* __restrict__ h0, const float* __restrict__ h1,
                      const float* __restrict__ h2, bf16_t* __restrict__ out) {
  long i = (long)blockIdx.x * 256 + threadIdx.x;
  long total = (long)B_ * S_ * 3 * D_;
  if (i >= total) return;
  long bs = i / (3 * D_);
  int  j  = (int)(i % (3 * D_));
  const float* src = (j < D_) ? h0 : (j < 2 * D_ ? h1 : h2);
  int jj = (j < D_) ? j : (j < 2 * D_ ? j - D_ : j - 2 * D_);
  out[i] = (bf16_t)src[bs * D_ + jj];
}

__global__ __launch_bounds__(256)
void noise_gather_kernel(const int* __restrict__ ids, const int* __restrict__ anchor,
                         const bool* __restrict__ keep, const float* __restrict__ embed,
                         bf16_t* __restrict__ out) {
  long i = (long)blockIdx.x * 256 + threadIdx.x;
  long total = (long)B_ * N_ * BS_ * D_;
  if (i >= total) return;
  int  d   = (int)(i & (D_ - 1));
  long row = i >> 10;
  int  m   = (int)(row & (BS_ - 1));
  int  bn  = (int)(row >> 4);
  int  tok = V_ - 1;                      // MASK_ID
  if (m == 0 && keep[bn]) {
    int b = bn / N_;
    int a = anchor[bn];
    a = a < 0 ? 0 : (a > S_ - 1 ? S_ - 1 : a);
    tok = ids[b * S_ + a];
  }
  out[i] = (bf16_t)embed[(size_t)tok * D_ + d];
}

// RoPE (fp32 in) -> bf16 out.  draftMode: pos = anchor[row/16] + row%16, else pos = row%S.
__global__ __launch_bounds__(256)
void rope_bf16_kernel(const float* __restrict__ in, bf16_t* __restrict__ out,
                      const int* __restrict__ anchor, int draftMode, long nrows) {
  long i = (long)blockIdx.x * 256 + threadIdx.x;
  long total = nrows * H_ * (HD_ / 2);
  if (i >= total) return;
  int  d = (int)(i & 63);
  long t = i >> 6;
  int  h = (int)(t & (H_ - 1));
  long row = t >> 3;
  int pos = draftMode ? (anchor[row >> 4] + (int)(row & 15)) : (int)(row % S_);
  float inv = __expf(-(float)d * (9.210340371976184f / 64.0f)); // 10000^(-d/64)
  float ang = (float)pos * inv;
  float sn, cs; __sincosf(ang, &sn, &cs);
  size_t base = (size_t)row * D_ + (size_t)h * HD_;
  float x1 = in[base + d], x2 = in[base + 64 + d];
  out[base + d]      = (bf16_t)(x1 * cs - x2 * sn);
  out[base + 64 + d] = (bf16_t)(x1 * sn + x2 * cs);
}

// v_ctx fp32 (B,S,H,HD) -> bf16 transposed (B,H,HD,S) so p@v B-fragments are contiguous.
__global__ __launch_bounds__(256)
void vctxT_kernel(const float* __restrict__ in, bf16_t* __restrict__ out) {
  long i = (long)blockIdx.x * 256 + threadIdx.x;
  long total = (long)B_ * S_ * D_;
  if (i >= total) return;
  int  d0 = (int)(i & (D_ - 1));
  long bs = i >> 10;
  int b = (int)(bs / S_), s = (int)(bs % S_);
  int h = d0 >> 7, d = d0 & 127;
  out[(((size_t)b * H_ + h) * HD_ + d) * S_ + s] = (bf16_t)in[i];
}

// v_dr fp32 (B,N,BS,H,HD) -> bf16 (B,N,H,HD,32) with keys 16..31 zero padded.
__global__ __launch_bounds__(256)
void vdrT_kernel(const float* __restrict__ in, bf16_t* __restrict__ out) {
  long i = (long)blockIdx.x * 256 + threadIdx.x;
  long total = (long)B_ * N_ * H_ * HD_ * 32;
  if (i >= total) return;
  int key = (int)(i & 31);
  long t  = i >> 5;
  int d   = (int)(t & 127);
  long t2 = t >> 7;
  int h   = (int)(t2 & (H_ - 1));
  int bn  = (int)(t2 >> 3);
  float v = 0.f;
  if (key < BS_) v = in[((size_t)bn * BS_ + key) * D_ + (size_t)h * HD_ + d];
  out[i] = (bf16_t)v;
}

// ---------------------------------------------------------------------------
// bf16 WMMA GEMM: C(MxN) = A(MxK) @ BT(NxK)^T, fp32 accumulate.
// 256 threads = 8 waves (4x2 over a 64x64 tile), BK=32 = one WMMA K-step.
// Double-buffered LDS tiles, filled with async global->LDS copies: issue
// tile t+1, s_wait_asynccnt for tile t, barrier, 2x WMMA, barrier.
// ---------------------------------------------------------------------------
#define GBM 64
#define GBN 64
#define GBK 32
__global__ __launch_bounds__(256)
void gemm_bf16_kernel(const bf16_t* __restrict__ A, const bf16_t* __restrict__ BT,
                      float* __restrict__ Cf, bf16_t* __restrict__ Cb,
                      int M, int Nn, int K) {
  __shared__ bf16_t As[2][GBM][GBK + 8];
  __shared__ bf16_t Bs[2][GBN][GBK + 8];  // Bs[n][k], contiguous in K (BT is NxK)
  const int tid  = threadIdx.x;
  const int lane = tid & 31, w = tid >> 5;
  const int wm = w & 3, wn = w >> 2;
  const bool hiL = lane >= 16; const int lh = lane & 15;
  const int kb = hiL ? 8 : 0, ks0 = hiL ? 16 : 0;
  const int rowBase = blockIdx.y * GBM, colBase = blockIdx.x * GBN;
  (void)M;

  const int rt = tid >> 2, ct = (tid & 3) * 8;   // 64 rows x 32 k, 16B per thread
  const bf16_t* aSrc = A  + (size_t)(rowBase + rt) * K + ct;
  const bf16_t* bSrc = BT + (size_t)(colBase + rt) * K + ct;

  v8f acc0 = {}, acc1 = {};
  // prologue: tile 0 in flight
  async_copy_b128(aSrc, &As[0][rt][ct]);
  async_copy_b128(bSrc, &Bs[0][rt][ct]);

  const int nk = K / GBK;
  for (int t = 0; t < nk; ++t) {
    const int buf = t & 1;
    if (t + 1 < nk) {
      const int k0 = (t + 1) * GBK;
      async_copy_b128(aSrc + k0, &As[buf ^ 1][rt][ct]);
      async_copy_b128(bSrc + k0, &Bs[buf ^ 1][rt][ct]);
      WAIT_ASYNC(2);          // tile t complete; tile t+1 still in flight
    } else {
      WAIT_ASYNC(0);
    }
    __syncthreads();
    bfu a, b0, b1;
    const int mrow = wm * 16 + lh;
    a.h[0] = *(const v8bf*)&As[buf][mrow][ks0];
    a.h[1] = *(const v8bf*)&As[buf][mrow][ks0 + 8];
    const int n0 = wn * 32 + lh, n1 = n0 + 16;
    b0.h[0] = *(const v8bf*)&Bs[buf][n0][ks0];
    b0.h[1] = *(const v8bf*)&Bs[buf][n0][ks0 + 8];
    b1.h[0] = *(const v8bf*)&Bs[buf][n1][ks0];
    b1.h[1] = *(const v8bf*)&Bs[buf][n1][ks0 + 8];
    acc0 = wmma_bf16(a.v, b0.v, acc0);
    acc1 = wmma_bf16(a.v, b1.v, acc1);
    __syncthreads();
  }
  #pragma unroll
  for (int r = 0; r < 8; ++r) {
    const int row = rowBase + wm * 16 + r + (hiL ? 8 : 0);
    const int c0  = colBase + wn * 32 + lh;
    if (Cf) {
      Cf[(size_t)row * Nn + c0]      = acc0[r];
      Cf[(size_t)row * Nn + c0 + 16] = acc1[r];
    } else {
      Cb[(size_t)row * Nn + c0]      = (bf16_t)acc0[r];
      Cb[(size_t)row * Nn + c0 + 16] = (bf16_t)acc1[r];
    }
  }
}

// ---------------------------------------------------------------------------
// Flash attention, one wave per (b,h,n).  q tile 16x128 preloaded as A-frags;
// scores S = q @ k^T in 16x32 chunks (8 WMMA), online softmax, p via LDS
// (s_wait_dscnt), o += p @ v with v pre-transposed (contiguous B-frags).
// ---------------------------------------------------------------------------
__device__ __forceinline__ void flash_update(
    int lane, const float* sm0, const float* sm1,
    float* mi, float* li, v8f* oac,
    bf16_t (*pT)[40],
    const bf16_t* __restrict__ vcol, size_t vstride, int vofs) {
  const bool hiL = lane >= 16; const int lh = lane & 15; const int kb = hiL ? 8 : 0;
  #pragma unroll
  for (int r = 0; r < 8; ++r) {
    float cm = fmaxf(sm0[r], sm1[r]);
    #pragma unroll
    for (int o = 8; o; o >>= 1) cm = fmaxf(cm, __shfl_xor(cm, o, 32));
    const float mn = fmaxf(mi[r], cm);
    const float alpha = __expf(mi[r] - mn);
    const float p0 = __expf(sm0[r] - mn);
    const float p1 = __expf(sm1[r] - mn);
    float rs = p0 + p1;
    #pragma unroll
    for (int o = 8; o; o >>= 1) rs += __shfl_xor(rs, o, 32);
    li[r] = li[r] * alpha + rs;
    mi[r] = mn;
    #pragma unroll
    for (int dt = 0; dt < 8; ++dt) oac[dt][r] *= alpha;
    const int m = r + (hiL ? 8 : 0);
    pT[m][lh]      = (bf16_t)p0;
    pT[m][16 + lh] = (bf16_t)p1;
  }
  asm volatile("s_wait_dscnt 0" ::: "memory");  // LDS p tile visible to frag reads
  bfu pa;
  pa.h[0] = *(const v8bf*)&pT[lh][2 * kb];
  pa.h[1] = *(const v8bf*)&pT[lh][2 * kb + 8];
  #pragma unroll
  for (int dt = 0; dt < 8; ++dt) {
    const bf16_t* vr = vcol + (size_t)(dt * 16 + lh) * vstride + vofs;
    bfu bv;
    bv.h[0] = *(const v8bf*)(vr);
    bv.h[1] = *(const v8bf*)(vr + 8);
    oac[dt] = wmma_bf16(pa.v, bv.v, oac[dt]);
  }
}

__global__ __launch_bounds__(32)
void attn_kernel(const bf16_t* __restrict__ qm, const bf16_t* __restrict__ kctx,
                 const bf16_t* __restrict__ vctxT, const bf16_t* __restrict__ kdr,
                 const bf16_t* __restrict__ vdrT,
                 const int* __restrict__ anchor, const bool* __restrict__ keep,
                 bf16_t* __restrict__ out) {
  const int n = blockIdx.x, h = blockIdx.y, b = blockIdx.z;
  const int bn = b * N_ + n;
  const int lane = threadIdx.x;
  const bool hiL = lane >= 16; const int lh = lane & 15;
  const int kb = hiL ? 8 : 0, ks0 = hiL ? 16 : 0;
  const int anc = anchor[bn];
  const bool kp = keep[bn];
  const float scale = 0.08838834764831845f;  // 1/sqrt(128)

  __shared__ bf16_t pT[16][40];

  bfu qf[4];
  {
    const bf16_t* qrow = qm + ((size_t)bn * BS_ + lh) * D_ + (size_t)h * HD_;
    #pragma unroll
    for (int kk = 0; kk < 4; ++kk) {
      qf[kk].h[0] = *(const v8bf*)(qrow + kk * 32 + ks0);
      qf[kk].h[1] = *(const v8bf*)(qrow + kk * 32 + ks0 + 8);
    }
  }

  float mi[8], li[8]; v8f oac[8];
  v8f zero = {};
  #pragma unroll
  for (int r = 0; r < 8; ++r) { mi[r] = -3.0e38f; li[r] = 0.f; }
  #pragma unroll
  for (int dt = 0; dt < 8; ++dt) oac[dt] = zero;

  const bf16_t* vcb = vctxT + ((size_t)b * H_ + h) * HD_ * (size_t)S_;
  const int climit = kp ? anc : 0;
  for (int base = 0; base < climit; base += 32) {
    v8f s0 = {}, s1 = {};
    const bf16_t* k0 = kctx + ((size_t)b * S_ + base + lh) * D_ + (size_t)h * HD_;
    const bf16_t* k1 = k0 + (size_t)16 * D_;
    #pragma unroll
    for (int kk = 0; kk < 4; ++kk) {
      bfu bb;
      const int ko = kk * 32 + ks0;
      bb.h[0] = *(const v8bf*)(k0 + ko); bb.h[1] = *(const v8bf*)(k0 + ko + 8);
      s0 = wmma_bf16(qf[kk].v, bb.v, s0);
      bb.h[0] = *(const v8bf*)(k1 + ko); bb.h[1] = *(const v8bf*)(k1 + ko + 8);
      s1 = wmma_bf16(qf[kk].v, bb.v, s1);
    }
    float sm0[8], sm1[8];
    const int key0 = base + lh, key1 = key0 + 16;
    #pragma unroll
    for (int r = 0; r < 8; ++r) {
      sm0[r] = (key0 < anc) ? s0[r] * scale : NEGF;
      sm1[r] = (key1 < anc) ? s1[r] * scale : NEGF;
    }
    flash_update(lane, sm0, sm1, mi, li, oac, pT, vcb, (size_t)S_, base + ks0);
  }
  { // draft block: 16 real keys, tile1 padded to exact-zero probability
    v8f s0 = {};
    const bf16_t* k0 = kdr + ((size_t)bn * BS_ + lh) * D_ + (size_t)h * HD_;
    #pragma unroll
    for (int kk = 0; kk < 4; ++kk) {
      bfu bb;
      const int ko = kk * 32 + ks0;
      bb.h[0] = *(const v8bf*)(k0 + ko); bb.h[1] = *(const v8bf*)(k0 + ko + 8);
      s0 = wmma_bf16(qf[kk].v, bb.v, s0);
    }
    float sm0[8], sm1[8];
    #pragma unroll
    for (int r = 0; r < 8; ++r) {
      sm0[r] = kp ? s0[r] * scale : NEGF;
      sm1[r] = -1.0e30f;
    }
    const bf16_t* vdb = vdrT + ((size_t)bn * H_ + h) * HD_ * 32;
    flash_update(lane, sm0, sm1, mi, li, oac, pT, vdb, (size_t)32, ks0);
  }
  #pragma unroll
  for (int dt = 0; dt < 8; ++dt) {
    #pragma unroll
    for (int r = 0; r < 8; ++r) {
      const int m = r + (hiL ? 8 : 0);
      const int d = dt * 16 + lh;
      out[((size_t)bn * BS_ + m) * D_ + (size_t)h * HD_ + d] = (bf16_t)(oac[dt][r] / li[r]);
    }
  }
}

// ---------------------------------------------------------------------------
// RMSNorm (fp32 in, bf16 out), one block per row.
// ---------------------------------------------------------------------------
__global__ __launch_bounds__(256)
void rmsnorm_kernel(const float* __restrict__ in, const float* __restrict__ w,
                    bf16_t* __restrict__ out) {
  const int row = blockIdx.x;
  const float* x = in + (size_t)row * D_;
  const int tid = threadIdx.x;
  float ss = 0.f;
  for (int i = tid; i < D_; i += 256) { float v = x[i]; ss += v * v; }
  #pragma unroll
  for (int o = 16; o; o >>= 1) ss += __shfl_xor(ss, o, 32);
  __shared__ float red[8];
  __shared__ float sscale;
  if ((tid & 31) == 0) red[tid >> 5] = ss;
  __syncthreads();
  if (tid == 0) {
    float t = 0.f;
    #pragma unroll
    for (int i = 0; i < 8; ++i) t += red[i];
    sscale = rsqrtf(t / (float)D_ + 1e-6f);
  }
  __syncthreads();
  const float sc = sscale;
  for (int i = tid; i < D_; i += 256) out[(size_t)row * D_ + i] = (bf16_t)(x[i] * sc * w[i]);
}

// ---------------------------------------------------------------------------
// Fused logits + online logsumexp + target capture + weighted NLL.
// One block per (b,n): 16 rows (BS) x V, streaming lm_head (L2 resident).
// ---------------------------------------------------------------------------
__global__ __launch_bounds__(256)
void loss_kernel(const bf16_t* __restrict__ hid, const bf16_t* __restrict__ lm,
                 const int* __restrict__ ids, const float* __restrict__ lmask,
                 const int* __restrict__ anchor, const bool* __restrict__ keep,
                 float* __restrict__ acc) {
  const int bn = blockIdx.x;
  const int b  = bn / N_;
  __shared__ bf16_t Hs[16][1032];  // padded stride -> conflict-free A-frag reads
  __shared__ int   tgt[16];
  __shared__ float wgt[16];
  __shared__ float tlog[16];
  __shared__ float redM[8][16];
  __shared__ float redS[8][16];
  const int tid = threadIdx.x, lane = tid & 31, w = tid >> 5;
  const bool hiL = lane >= 16; const int lh = lane & 15;
  const int kb = hiL ? 8 : 0, ks0 = hiL ? 16 : 0;

  for (int c = tid; c < 16 * 128; c += 256) {
    int row = c >> 7, cc = (c & 127) * 8;
    async_copy_b128(hid + ((size_t)bn * 16 + row) * D_ + cc, &Hs[row][cc]);
  }
  WAIT_ASYNC(0);
  if (tid < 16) {
    const int m = tid;
    const int a = anchor[bn];
    const int label = a + m;
    const bool valid = label < S_;
    int safe = label < S_ - 1 ? label : S_ - 1;
    if (safe < 0) safe = 0;
    tgt[m] = ids[b * S_ + safe];
    float wv = 0.f;
    if (keep[bn] && valid && m > 0)
      wv = lmask[b * S_ + safe] * __expf(-(float)(m - 1) / 7.0f);
    wgt[m] = wv;
    tlog[m] = 0.f;
  }
  __syncthreads();

  float rmax[8], rsum[8];
  #pragma unroll
  for (int r = 0; r < 8; ++r) { rmax[r] = -3.0e38f; rsum[r] = 0.f; }

  for (int vb = 0; vb < V_; vb += 128) {   // 250 iterations, exact
    const int col = vb + w * 16 + lh;
    const bf16_t* lrow = lm + (size_t)col * D_;
    if (vb + 128 < V_) __builtin_prefetch(lrow + (size_t)128 * D_, 0, 1);
    v8f a8 = {};
    #pragma unroll 4
    for (int ksi = 0; ksi < D_; ksi += 32) {
      bfu a, bb;
      a.h[0]  = *(const v8bf*)&Hs[lh][ksi + ks0];
      a.h[1]  = *(const v8bf*)&Hs[lh][ksi + ks0 + 8];
      bb.h[0] = *(const v8bf*)(lrow + ksi + ks0);
      bb.h[1] = *(const v8bf*)(lrow + ksi + ks0 + 8);
      a8 = wmma_bf16(a.v, bb.v, a8);
    }
    #pragma unroll
    for (int r = 0; r < 8; ++r) {
      const float x = a8[r];
      const int rowm = r + (hiL ? 8 : 0);
      if (col == tgt[rowm]) tlog[rowm] = x;
      if (x > rmax[r]) { rsum[r] = rsum[r] * __expf(rmax[r] - x) + 1.f; rmax[r] = x; }
      else             { rsum[r] += __expf(x - rmax[r]); }
    }
  }
  #pragma unroll
  for (int r = 0; r < 8; ++r) {
    #pragma unroll
    for (int o = 8; o; o >>= 1) {
      const float om = __shfl_xor(rmax[r], o, 32);
      const float os = __shfl_xor(rsum[r], o, 32);
      const float mn = fmaxf(rmax[r], om);
      rsum[r] = rsum[r] * __expf(rmax[r] - mn) + os * __expf(om - mn);
      rmax[r] = mn;
    }
  }
  if (lh == 0) {
    #pragma unroll
    for (int r = 0; r < 8; ++r) {
      const int rowm = r + (hiL ? 8 : 0);
      redM[w][rowm] = rmax[r];
      redS[w][rowm] = rsum[r];
    }
  }
  __syncthreads();
  if (tid < 16) {
    const int m = tid;
    float M = -3.0e38f;
    for (int wv = 0; wv < 8; ++wv) M = fmaxf(M, redM[wv][m]);
    float Ssum = 0.f;
    for (int wv = 0; wv < 8; ++wv) Ssum += redS[wv][m] * __expf(redM[wv][m] - M);
    const float logZ = M + __logf(Ssum);
    const float lpt = logZ - tlog[m];
    redM[0][m] = lpt * wgt[m];
    redS[0][m] = wgt[m];
  }
  __syncthreads();
  if (tid == 0) {
    float c = 0.f, wsum = 0.f;
    for (int m = 0; m < 16; ++m) { c += redM[0][m]; wsum += redS[0][m]; }
    atomicAdd(&acc[0], c);
    atomicAdd(&acc[1], wsum);
  }
}

__global__ void finalize_kernel(const float* __restrict__ acc, float* __restrict__ out) {
  out[0] = acc[0] / fmaxf(acc[1], 1.0f);
}

// ---------------------------------------------------------------------------
// Host orchestration
// ---------------------------------------------------------------------------
extern "C" void kernel_launch(void* const* d_in, const int* in_sizes, int n_in,
                              void* d_out, int out_size, void* d_ws, size_t ws_size,
                              hipStream_t stream) {
  (void)in_sizes; (void)n_in; (void)out_size; (void)ws_size;
  const int*   input_ids = (const int*)d_in[0];
  const float* hs0       = (const float*)d_in[1];
  const float* hs1       = (const float*)d_in[2];
  const float* hs2       = (const float*)d_in[3];
  const float* loss_mask = (const float*)d_in[4];
  const float* lm_head   = (const float*)d_in[5];
  const int*   anchor    = (const int*)d_in[6];
  const bool*  keep      = (const bool*)d_in[7];
  const float* embed     = (const float*)d_in[8];
  const float* W_ctx     = (const float*)d_in[9];
  const float* Wq        = (const float*)d_in[10];
  const float* Wk        = (const float*)d_in[11];
  const float* Wv        = (const float*)d_in[12];
  const float* Wo        = (const float*)d_in[13];
  const float* norm_w    = (const float*)d_in[14];

  char* ws = (char*)d_ws; size_t off = 0;
  auto alloc = [&](size_t bytes) -> void* {
    void* p = ws + off; off = (off + bytes + 255) & ~(size_t)255; return p;
  };
  const long RS = (long)B_ * S_;          // 4096 ctx rows
  const long RD = (long)B_ * N_ * BS_;    // 4096 draft rows

  bf16_t* lm_bf    = (bf16_t*)alloc((size_t)V_ * D_ * 2);     // (V,D) already NxK
  bf16_t* wctxT    = (bf16_t*)alloc((size_t)3 * D_ * D_ * 2); // (D, 3D)
  bf16_t* wqT      = (bf16_t*)alloc((size_t)D_ * D_ * 2);
  bf16_t* wkT      = (bf16_t*)alloc((size_t)D_ * D_ * 2);
  bf16_t* wvT      = (bf16_t*)alloc((size_t)D_ * D_ * 2);
  bf16_t* woT      = (bf16_t*)alloc((size_t)D_ * D_ * 2);
  bf16_t* hscat    = (bf16_t*)alloc((size_t)RS * 3 * D_ * 2);
  bf16_t* ctx_bf   = (bf16_t*)alloc((size_t)RS * D_ * 2);
  float*  tmpf     = (float*) alloc((size_t)RS * D_ * 4);  // reused fp32 GEMM output
  bf16_t* kctx_bf  = (bf16_t*)alloc((size_t)RS * D_ * 2);
  bf16_t* vctxT    = (bf16_t*)alloc((size_t)RS * D_ * 2);
  bf16_t* noise_bf = (bf16_t*)alloc((size_t)RD * D_ * 2);
  bf16_t* q_bf     = (bf16_t*)alloc((size_t)RD * D_ * 2);
  bf16_t* kdr_bf   = (bf16_t*)alloc((size_t)RD * D_ * 2);
  bf16_t* vdrT     = (bf16_t*)alloc((size_t)B_ * N_ * H_ * HD_ * 32 * 2);
  bf16_t* attn_bf  = (bf16_t*)alloc((size_t)RD * D_ * 2);
  bf16_t* hid_bf   = (bf16_t*)alloc((size_t)RD * D_ * 2);
  float*  acc      = (float*) alloc(2 * sizeof(float));

  (void)hipMemsetAsync(acc, 0, 2 * sizeof(float), stream);
  auto g1 = [](long n) { return dim3((unsigned)((n + 255) / 256)); };

  // bf16 weight copies (GEMM B operands pre-transposed to NxK)
  convert_bf16_kernel<<<g1((long)V_ * D_), 256, 0, stream>>>(lm_head, lm_bf, (long)V_ * D_);
  convert_T_bf16_kernel<<<g1((long)3 * D_ * D_), 256, 0, stream>>>(W_ctx, wctxT, 3 * D_, D_);
  convert_T_bf16_kernel<<<g1((long)D_ * D_), 256, 0, stream>>>(Wq, wqT, D_, D_);
  convert_T_bf16_kernel<<<g1((long)D_ * D_), 256, 0, stream>>>(Wk, wkT, D_, D_);
  convert_T_bf16_kernel<<<g1((long)D_ * D_), 256, 0, stream>>>(Wv, wvT, D_, D_);
  convert_T_bf16_kernel<<<g1((long)D_ * D_), 256, 0, stream>>>(Wo, woT, D_, D_);

  // ctx = [hs0|hs1|hs2] @ W_ctx
  concat_hs_kernel<<<g1(RS * 3 * D_), 256, 0, stream>>>(hs0, hs1, hs2, hscat);
  gemm_bf16_kernel<<<dim3(D_ / GBN, (unsigned)(RS / GBM)), 256, 0, stream>>>(
      hscat, wctxT, nullptr, ctx_bf, (int)RS, D_, 3 * D_);

  // k_ctx = rope(ctx @ Wk), v_ctxT = transpose(ctx @ Wv)
  gemm_bf16_kernel<<<dim3(D_ / GBN, (unsigned)(RS / GBM)), 256, 0, stream>>>(
      ctx_bf, wkT, tmpf, nullptr, (int)RS, D_, D_);
  rope_bf16_kernel<<<g1(RS * H_ * 64), 256, 0, stream>>>(tmpf, kctx_bf, nullptr, 0, RS);
  gemm_bf16_kernel<<<dim3(D_ / GBN, (unsigned)(RS / GBM)), 256, 0, stream>>>(
      ctx_bf, wvT, tmpf, nullptr, (int)RS, D_, D_);
  vctxT_kernel<<<g1(RS * D_), 256, 0, stream>>>(tmpf, vctxT);

  // draft tokens: q, k_dr, v_drT
  noise_gather_kernel<<<g1(RD * D_), 256, 0, stream>>>(input_ids, anchor, keep, embed, noise_bf);
  gemm_bf16_kernel<<<dim3(D_ / GBN, (unsigned)(RD / GBM)), 256, 0, stream>>>(
      noise_bf, wqT, tmpf, nullptr, (int)RD, D_, D_);
  rope_bf16_kernel<<<g1(RD * H_ * 64), 256, 0, stream>>>(tmpf, q_bf, anchor, 1, RD);
  gemm_bf16_kernel<<<dim3(D_ / GBN, (unsigned)(RD / GBM)), 256, 0, stream>>>(
      noise_bf, wkT, tmpf, nullptr, (int)RD, D_, D_);
  rope_bf16_kernel<<<g1(RD * H_ * 64), 256, 0, stream>>>(tmpf, kdr_bf, anchor, 1, RD);
  gemm_bf16_kernel<<<dim3(D_ / GBN, (unsigned)(RD / GBM)), 256, 0, stream>>>(
      noise_bf, wvT, tmpf, nullptr, (int)RD, D_, D_);
  vdrT_kernel<<<g1((long)B_ * N_ * H_ * HD_ * 32), 256, 0, stream>>>(tmpf, vdrT);

  // flash attention
  attn_kernel<<<dim3(N_, H_, B_), 32, 0, stream>>>(
      q_bf, kctx_bf, vctxT, kdr_bf, vdrT, anchor, keep, attn_bf);

  // hid = RMSNorm(attn @ Wo)
  gemm_bf16_kernel<<<dim3(D_ / GBN, (unsigned)(RD / GBM)), 256, 0, stream>>>(
      attn_bf, woT, tmpf, nullptr, (int)RD, D_, D_);
  rmsnorm_kernel<<<dim3((unsigned)RD), 256, 0, stream>>>(tmpf, norm_w, hid_bf);

  // fused logits / logsumexp / weighted NLL
  loss_kernel<<<dim3(B_ * N_), 256, 0, stream>>>(
      hid_bf, lm_bf, input_ids, loss_mask, anchor, keep, acc);
  finalize_kernel<<<1, 1, 0, stream>>>(acc, (float*)d_out);
}
